// CosineGating_74629351735465
// MI455X (gfx1250) — compile-verified, matrix operand order checked by reading, not run on hardware
//
#include <hip/hip_runtime.h>
#include <stdint.h>

typedef __attribute__((ext_vector_type(16))) _Float16 v16h;
typedef __attribute__((ext_vector_type(2)))  float    v2f;
typedef __attribute__((ext_vector_type(8)))  float    v8f;

#define N_TOK  65536
#define D_IN   2048
#define EMB_D  256
#define NEXP   64
#define KC     32      // K-chunk per wmma f16 step
#define MTILE  64      // token rows per workgroup
#define EPSF   1e-12f

__global__ __launch_bounds__(256)
void cosine_gating_fused(const float* __restrict__ x,
                         const float* __restrict__ Wp,
                         const float* __restrict__ emb,
                         const float* __restrict__ temp_p,
                         float* __restrict__ out_ew,
                         int*   __restrict__ out_ti,
                         float* __restrict__ out_lg,
                         float* __restrict__ out_cos,
                         float* __restrict__ out_rp)
{
    // GEMM1 staging (sA 4KB + sB 16KB) and GEMM2 B-matrix (sEmbn 64KB)
    // live in disjoint phases -> alias one 64KB region.
    __shared__ __align__(16) unsigned char smemU[64 * 1024];
    __shared__ __align__(16) float sProj[MTILE * EMB_D];   // 64 KB fp32 projection
    __shared__ float sCos[MTILE * NEXP];                   // 16 KB cosine tile
    __shared__ float sEinv[NEXP];
    __shared__ float sRinv[MTILE];
    __shared__ float sM1[MTILE], sM2[MTILE], sRsum[MTILE];
    __shared__ int   sI1[MTILE], sI2[MTILE];

    _Float16* sA    = (_Float16*)smemU;                    // [64][32] f16
    _Float16* sB    = (_Float16*)(smemU + 4096);           // [256][32] f16 (col-major-K)
    float*    sEmbn = (float*)smemU;                       // [64][256] f32, after GEMM1

    const int tid  = threadIdx.x;
    const int lane = tid & 31;
    const int wave = tid >> 5;          // 0..7
    const int half = lane >> 4;         // K sub-pattern select
    const int l16  = lane & 15;
    const int row0 = blockIdx.x * MTILE;
    const float tval = temp_p[0];

    // ---- expert-embedding column inverse L2 norms ----
    if (tid < NEXP) {
        float s = 0.f;
        for (int k = 0; k < EMB_D; ++k) {
            float v = emb[k * NEXP + tid];
            s += v * v;
        }
        sEinv[tid] = rsqrtf(s + EPSF);
    }

    // =========== GEMM1: proj[64x256] = x[64x2048] @ Wp[2048x256], f16 WMMA ===========
    const int col0 = wave * 32;         // this wave's two 16-col tiles
    const int col1 = wave * 32 + 16;
    const uint4* a4 = (const uint4*)sA; // row r = 4 uint4
    const uint4* b4 = (const uint4*)sB; // col c = 4 uint4

    // staging coordinates
    const int ar = tid >> 2;            // A row 0..63
    const int ak = (tid & 3) * 8;       // A k-base (8 elements)
    const int bc = (tid & 63) * 4;      // B col-base (4 cols)
    const int br = (tid >> 6) * 8;      // B k-base (8 rows)

    v8f acc[4][2] = {};                 // 4 row-tiles x 2 col-tiles

    for (int k0 = 0; k0 < D_IN; k0 += KC) {
        // stage A: 64x32 fp32 -> f16, one b128 store per thread
        {
            const float* src = x + (size_t)(row0 + ar) * D_IN + k0 + ak;
            float4 f0 = ((const float4*)src)[0];
            float4 f1 = ((const float4*)src)[1];
            union { uint4 q; _Float16 h[8]; } pk;
            pk.h[0] = (_Float16)f0.x; pk.h[1] = (_Float16)f0.y;
            pk.h[2] = (_Float16)f0.z; pk.h[3] = (_Float16)f0.w;
            pk.h[4] = (_Float16)f1.x; pk.h[5] = (_Float16)f1.y;
            pk.h[6] = (_Float16)f1.z; pk.h[7] = (_Float16)f1.w;
            *(uint4*)(sA + ar * KC + ak) = pk.q;
            if (k0 + KC < D_IN)
                __builtin_prefetch((const char*)src + KC * sizeof(float), 0, 0);
        }
        // stage B transposed: 8 coalesced float4 row-reads, pack per-column, b128 stores
        {
            float w[8][4];
            #pragma unroll
            for (int j = 0; j < 8; ++j)
                *(float4*)w[j] = *(const float4*)(Wp + (size_t)(k0 + br + j) * EMB_D + bc);
            #pragma unroll
            for (int c = 0; c < 4; ++c) {
                union { uint4 q; _Float16 h[8]; } pk;
                #pragma unroll
                for (int j = 0; j < 8; ++j) pk.h[j] = (_Float16)w[j][c];
                *(uint4*)(sB + (bc + c) * KC + br) = pk.q;
            }
        }
        __syncthreads();

        // fragments: two ds_load_b128 each (documented 16-bit 16x32 layout)
        union { v16h v; uint4 q[2]; } fb0, fb1, fa[4];
        fb0.q[0] = b4[(col0 + l16) * 4 + half];
        fb0.q[1] = b4[(col0 + l16) * 4 + half + 2];
        fb1.q[0] = b4[(col1 + l16) * 4 + half];
        fb1.q[1] = b4[(col1 + l16) * 4 + half + 2];
        #pragma unroll
        for (int rt = 0; rt < 4; ++rt) {
            int r = rt * 16 + l16;
            fa[rt].q[0] = a4[r * 4 + half];
            fa[rt].q[1] = a4[r * 4 + half + 2];
        }
        #pragma unroll
        for (int rt = 0; rt < 4; ++rt) {
            acc[rt][0] = __builtin_amdgcn_wmma_f32_16x16x32_f16(false, fa[rt].v, false, fb0.v,
                                                                (short)0, acc[rt][0], false, false);
            acc[rt][1] = __builtin_amdgcn_wmma_f32_16x16x32_f16(false, fa[rt].v, false, fb1.v,
                                                                (short)0, acc[rt][1], false, false);
        }
        __syncthreads();
    }

    // spill accumulators to sProj (lane l16 = column, VGPR j = row j + 8*half)
    #pragma unroll
    for (int rt = 0; rt < 4; ++rt) {
        #pragma unroll
        for (int j = 0; j < 8; ++j) {
            int r = rt * 16 + j + 8 * half;
            sProj[r * EMB_D + col0 + l16] = acc[rt][0][j];
            sProj[r * EMB_D + col1 + l16] = acc[rt][1][j];
        }
    }
    __syncthreads();

    // ---- stage normalized expert embeddings transposed: sEmbn[e][k] (aliases sA/sB) ----
    #pragma unroll 8
    for (int i = 0; i < 64; ++i) {
        int flat = i * 256 + tid;       // coalesced read of emb[k][e]
        int k = flat >> 6, e = flat & 63;
        sEmbn[e * EMB_D + k] = emb[(size_t)k * NEXP + e] * sEinv[e];
    }
    // ---- row inverse L2 norms ----
    if (tid < MTILE) {
        const float4* pr = (const float4*)(sProj + tid * EMB_D);
        float s = 0.f;
        for (int c = 0; c < EMB_D / 4; ++c) {
            float4 v = pr[c];
            s += v.x * v.x + v.y * v.y + v.z * v.z + v.w * v.w;
        }
        sRinv[tid] = rsqrtf(s + EPSF);
    }
    __syncthreads();

    // =========== GEMM2: cos[64x64] = proj_n @ emb_n, fp32 WMMA (16x16x4) ===========
    {
        const int mt = wave & 3;              // row tile 0..3
        const int nb = (wave >> 2) * 32;      // expert base: 0 or 32 (two 16-col tiles)
        const int arow = mt * 16 + l16;
        v8f c0 = {}, c1 = {};
        #pragma unroll 4
        for (int s = 0; s < EMB_D / 4; ++s) {
            int kk = s * 4 + half * 2;        // even -> 8B-aligned b64 reads
            float2 av = *(const float2*)(sProj + arow * EMB_D + kk);
            float2 b0 = *(const float2*)(sEmbn + (nb + l16) * EMB_D + kk);
            float2 b1 = *(const float2*)(sEmbn + (nb + 16 + l16) * EMB_D + kk);
            v2f A;  A.x  = av.x; A.y  = av.y;
            v2f B0; B0.x = b0.x; B0.y = b0.y;
            v2f B1; B1.x = b1.x; B1.y = b1.y;
            c0 = __builtin_amdgcn_wmma_f32_16x16x4_f32(false, A, false, B0,
                                                       (short)0, c0, false, false);
            c1 = __builtin_amdgcn_wmma_f32_16x16x4_f32(false, A, false, B1,
                                                       (short)0, c1, false, false);
        }
        #pragma unroll
        for (int j = 0; j < 8; ++j) {
            int r = mt * 16 + j + 8 * half;
            float rv = sRinv[r];
            sCos[r * NEXP + nb + l16]      = c0[j] * rv;
            sCos[r * NEXP + nb + 16 + l16] = c1[j] * rv;
        }
    }
    __syncthreads();

    // ---- per-row top-2 + softmax scalars ----
    if (tid < MTILE) {
        const float* cr = sCos + tid * NEXP;
        float m1 = -3.4e38f, m2 = -3.4e38f;
        int   i1 = 0,        i2 = 0;
        for (int e = 0; e < NEXP; ++e) {
            float lg = cr[e] * tval;
            if (lg > m1)      { m2 = m1; i2 = i1; m1 = lg; i1 = e; }
            else if (lg > m2) { m2 = lg; i2 = e; }
        }
        float rs = 0.f;
        for (int e = 0; e < NEXP; ++e) rs += expf(cr[e] * tval - m1);
        sM1[tid] = m1; sM2[tid] = m2; sRsum[tid] = rs;
        sI1[tid] = i1; sI2[tid] = i2;
    }
    __syncthreads();

    // ---- coalesced output streaming ----
    #pragma unroll 4
    for (int q = 0; q < 16; ++q) {
        int idx = q * 256 + tid;              // 0..4095
        int row = idx >> 6;
        int e   = idx & 63;
        size_t g = (size_t)(row0 + row) * NEXP + e;
        float c   = sCos[row * NEXP + e];
        float lg  = c * tval;
        float m1  = sM1[row];
        float em2 = expf(sM2[row] - m1);
        float mden = 1.f + em2;               // exp(NEG_INF - m1) underflows to 0
        float ew = (e == sI1[row]) ? (1.f / mden)
                 : (e == sI2[row]) ? (em2 / mden) : 0.f;
        out_ew[g]  = ew;
        out_lg[g]  = lg;
        out_cos[g] = c;
        out_rp[g]  = expf(lg - m1) / sRsum[row];
    }
    if (tid < 2 * MTILE) {
        int row = tid >> 1;
        out_ti[(size_t)(row0 + row) * 2 + (tid & 1)] =
            (tid & 1) ? sI2[row] : sI1[row];
    }
}

extern "C" void kernel_launch(void* const* d_in, const int* in_sizes, int n_in,
                              void* d_out, int out_size, void* d_ws, size_t ws_size,
                              hipStream_t stream) {
    (void)in_sizes; (void)n_in; (void)out_size; (void)d_ws; (void)ws_size;
    const float* x    = (const float*)d_in[0];
    const float* Wp   = (const float*)d_in[1];
    const float* emb  = (const float*)d_in[2];
    const float* temp = (const float*)d_in[3];

    float* out = (float*)d_out;
    const size_t NE = (size_t)N_TOK * NEXP;
    float* out_ew  = out;                              // [N, E]
    int*   out_ti  = (int*)(out + NE);                 // [N, 2] int32
    float* out_lg  = out + NE + (size_t)N_TOK * 2;     // [N, E]
    float* out_cos = out_lg + NE;                      // [N, E]
    float* out_rp  = out_cos + NE;                     // [N, E]

    dim3 grid(N_TOK / MTILE);   // 1024 blocks
    dim3 block(256);            // 8 wave32
    cosine_gating_fused<<<grid, block, 0, stream>>>(x, Wp, emb, temp,
                                                    out_ew, out_ti, out_lg,
                                                    out_cos, out_rp);
}
